// MambaBlock_65369402245501
// MI455X (gfx1250) — matrix operand order repaired and checked by reading, not compiled, not run on hardware
//
#include <hip/hip_runtime.h>
#include <hip/hip_fp16.h>
#include <cmath>

#define BATCH   2
#define LSEQ    1024
#define D_MODEL 1024
#define D_INNER 2048
#define DT_RANK 64
#define D_STATE 16
#define ROWS    (BATCH * LSEQ)          // 2048 token rows
#define DBC_W   (DT_RANK + 2 * D_STATE) // 96

typedef __attribute__((ext_vector_type(16))) _Float16 v16h;
typedef __attribute__((ext_vector_type(8)))  _Float16 v8h;
typedef __attribute__((ext_vector_type(8)))  float    v8f;

// ---------------------------------------------------------------------------
// WMMA fragment loaders (contiguous k order in A and B, wave32)
// A: MxK f16 row-major. lane L holds row m0+(L&15); halves k0+16g+{0..7} and
//    k0+16g+8+{0..7} with g = L>>4  -> two 16B loads.
// B: weight W stored (N,K) row-major; B[k][n] = W[n][k]. lane L holds column
//    n0+(L&15) = row n of W, 16 contiguous halves at k0+16g -> one 32B load.
// ---------------------------------------------------------------------------
__device__ __forceinline__ v16h load_a_frag(const _Float16* __restrict__ A,
                                            int lda, int m0, int k0, int lane) {
  const int r = m0 + (lane & 15);
  const int g = lane >> 4;
  const _Float16* p = A + (size_t)r * lda + k0 + g * 16;
  v8h lo = *reinterpret_cast<const v8h*>(p);
  v8h hi = *reinterpret_cast<const v8h*>(p + 8);
  v16h out;
#pragma unroll
  for (int i = 0; i < 8; ++i) { out[i] = lo[i]; out[8 + i] = hi[i]; }
  return out;
}

__device__ __forceinline__ v16h load_b_frag(const _Float16* __restrict__ W,
                                            int ldb, int n0, int k0, int lane) {
  const int n = n0 + (lane & 15);
  const int g = lane >> 4;
  const _Float16* p = W + (size_t)n * ldb + k0 + g * 16;
  return *reinterpret_cast<const v16h*>(p);   // 32B contiguous
}

// ---------------------------------------------------------------------------
// Generic WMMA GEMM: C(MxN) = A(MxK) * W(N,K)^T [+ epilogue]
// One wave computes a (16*MT) x (16*NT) tile: A frags reused across N tiles,
// B frags reused across M tiles -> MT*NT WMMAs per (2*MT + 2*NT) b128 loads.
// EPI==1: C = softplus(acc + bias[n])
// ---------------------------------------------------------------------------
template <int MT, int NT, int EPI>
__global__ __launch_bounds__(256) void gemm_wmma_kernel(
    const _Float16* __restrict__ A, const _Float16* __restrict__ W,
    float* __restrict__ C, const float* __restrict__ bias,
    int M, int N, int K, int lda, int ldb, int ldc) {
  const int lane  = threadIdx.x & 31;
  const int gwave = blockIdx.x * (blockDim.x >> 5) + (threadIdx.x >> 5);
  const int ngrp  = N / (16 * NT);
  const int mt    = gwave / ngrp;
  if (mt * 16 * MT >= M) return;            // wave-uniform
  const int m0 = mt * 16 * MT;
  const int n0 = (gwave - mt * ngrp) * 16 * NT;

  v8f acc[MT][NT];
#pragma unroll
  for (int r = 0; r < MT; ++r)
#pragma unroll
    for (int t = 0; t < NT; ++t)
#pragma unroll
      for (int i = 0; i < 8; ++i) acc[r][t][i] = 0.f;

  for (int k0 = 0; k0 < K; k0 += 32) {
    v16h a[MT];
#pragma unroll
    for (int r = 0; r < MT; ++r)
      a[r] = load_a_frag(A, lda, m0 + 16 * r, k0, lane);
#pragma unroll
    for (int t = 0; t < NT; ++t) {
      const v16h b = load_b_frag(W, ldb, n0 + 16 * t, k0, lane);
#pragma unroll
      for (int r = 0; r < MT; ++r)
        acc[r][t] = __builtin_amdgcn_wmma_f32_16x16x32_f16(
            /*neg_a=*/false, a[r], /*neg_b=*/false, b,
            /*c_mod=*/(short)0, acc[r][t],
            /*reuse_a=*/false, /*reuse_b=*/false);
    }
  }

  const int cn = lane & 15;
  const int rb = (lane >> 4) * 8;
#pragma unroll
  for (int r = 0; r < MT; ++r) {
#pragma unroll
    for (int t = 0; t < NT; ++t) {
      const int col = n0 + 16 * t + cn;
      float bv = 0.f;
      if (EPI == 1) bv = bias[col];
#pragma unroll
      for (int v = 0; v < 8; ++v) {
        float val = acc[r][t][v];
        if (EPI == 1) {
          val += bv;
          val = (val > 20.f) ? val : log1pf(expf(val));   // softplus
        }
        C[(size_t)(m0 + 16 * r + rb + v) * ldc + col] = val;
      }
    }
  }
}

// ---------------------------------------------------------------------------
// Elementwise helpers
// ---------------------------------------------------------------------------
__global__ void cvt_f32_f16_kernel(const float* __restrict__ s,
                                   _Float16* __restrict__ d, int n) {
  int i = blockIdx.x * blockDim.x + threadIdx.x;
  if (i < n) d[i] = (_Float16)s[i];
}

__global__ void aneg_kernel(const float* __restrict__ A_log,
                            float* __restrict__ Aneg, int n) {
  int i = blockIdx.x * blockDim.x + threadIdx.x;
  if (i < n) Aneg[i] = -__expf(A_log[i]);
}

// Depthwise conv1d (k=3, same pad) + bias + SiLU over xi = xz[:, 0:D_INNER]
__global__ void conv_silu_kernel(const float* __restrict__ xz,
                                 const float* __restrict__ Wc,
                                 const float* __restrict__ bc,
                                 float* __restrict__ xc,
                                 _Float16* __restrict__ xch) {
  size_t idx = (size_t)blockIdx.x * blockDim.x + threadIdx.x;
  if (idx >= (size_t)ROWS * D_INNER) return;
  const int d = (int)(idx & (D_INNER - 1));
  const size_t row = idx >> 11;                     // token row (b*L + l)
  const int l = (int)(row & (LSEQ - 1));
  const float* base = xz + row * (2 * D_INNER) + d; // xi columns
  float acc = bc[d];
  if (l > 0)        acc += Wc[d * 3 + 0] * base[-(ptrdiff_t)(2 * D_INNER)];
  acc += Wc[d * 3 + 1] * base[0];
  if (l < LSEQ - 1) acc += Wc[d * 3 + 2] * base[2 * D_INNER];
  const float s = acc / (1.f + __expf(-acc));       // SiLU
  xc[idx] = s;
  xch[idx] = (_Float16)s;
}

// SSM update: h_new = exp(dt*A)*h + dt*B*xc ; y = <h_new,C> + D*xc ;
// emits yz = y * silu(z) as f16 (A operand of final GEMM), h_new -> d_out.
__global__ void ssm_kernel(const float* __restrict__ xz,
                           const float* __restrict__ xc,
                           const float* __restrict__ dbc,
                           const float* __restrict__ delta,
                           const float* __restrict__ Aneg,
                           const float* __restrict__ Dp,
                           const float* __restrict__ h,
                           float* __restrict__ h_new,
                           _Float16* __restrict__ yz) {
  size_t idx = (size_t)blockIdx.x * blockDim.x + threadIdx.x;
  if (idx >= (size_t)ROWS * D_INNER) return;
  const int d = (int)(idx & (D_INNER - 1));
  const size_t m = idx >> 11;
  const float dt = delta[idx];
  const float xv = xc[idx];
  const float* Bm = dbc + m * DBC_W + DT_RANK;
  const float* Cm = Bm + D_STATE;
  const float* Ar = Aneg + (size_t)d * D_STATE;
  const float4* h4 = reinterpret_cast<const float4*>(h + idx * D_STATE);
  float4* hn4 = reinterpret_cast<float4*>(h_new + idx * D_STATE);
  float y = 0.f;
#pragma unroll
  for (int q = 0; q < 4; ++q) {
    const float4 hv = h4[q];
    const float dB = dt * xv;
    float4 hn;
    hn.x = __expf(dt * Ar[4 * q + 0]) * hv.x + dB * Bm[4 * q + 0];
    hn.y = __expf(dt * Ar[4 * q + 1]) * hv.y + dB * Bm[4 * q + 1];
    hn.z = __expf(dt * Ar[4 * q + 2]) * hv.z + dB * Bm[4 * q + 2];
    hn.w = __expf(dt * Ar[4 * q + 3]) * hv.w + dB * Bm[4 * q + 3];
    hn4[q] = hn;
    y += hn.x * Cm[4 * q + 0] + hn.y * Cm[4 * q + 1] +
         hn.z * Cm[4 * q + 2] + hn.w * Cm[4 * q + 3];
  }
  y += Dp[d] * xv;
  const float zv = xz[m * (2 * D_INNER) + D_INNER + d];   // z half of xz
  const float sz = zv / (1.f + __expf(-zv));
  yz[idx] = (_Float16)(y * sz);
}

// ---------------------------------------------------------------------------
extern "C" void kernel_launch(void* const* d_in, const int* in_sizes, int n_in,
                              void* d_out, int out_size, void* d_ws, size_t ws_size,
                              hipStream_t stream) {
  const float* x      = (const float*)d_in[0];
  const float* h      = (const float*)d_in[1];
  const float* W_in   = (const float*)d_in[2];
  const float* W_conv = (const float*)d_in[3];
  const float* b_conv = (const float*)d_in[4];
  const float* W_x    = (const float*)d_in[5];
  const float* W_dt   = (const float*)d_in[6];
  const float* b_dt   = (const float*)d_in[7];
  const float* A_log  = (const float*)d_in[8];
  const float* D_par  = (const float*)d_in[9];
  const float* W_out  = (const float*)d_in[10];

  float* out_p  = (float*)d_out;                           // (B,L,D_MODEL)
  float* h_newp = (float*)d_out + (size_t)ROWS * D_MODEL;  // (B,L,D_INNER,16)

  char* ws = (char*)d_ws;
  size_t off = 0;
  auto alloc = [&](size_t bytes) -> char* {
    char* p = ws + off;
    off += (bytes + 255) & ~(size_t)255;
    return p;
  };
  _Float16* xh    = (_Float16*)alloc((size_t)ROWS * D_MODEL * 2);
  _Float16* Winh  = (_Float16*)alloc((size_t)2 * D_INNER * D_MODEL * 2);
  _Float16* Wxh   = (_Float16*)alloc((size_t)DBC_W * D_INNER * 2);
  _Float16* Wdth  = (_Float16*)alloc((size_t)D_INNER * DT_RANK * 2);
  _Float16* Wouth = (_Float16*)alloc((size_t)D_MODEL * D_INNER * 2);
  float*    xzb   = (float*)   alloc((size_t)ROWS * 2 * D_INNER * 4);
  float*    xcb   = (float*)   alloc((size_t)ROWS * D_INNER * 4);
  _Float16* xch   = (_Float16*)alloc((size_t)ROWS * D_INNER * 2);
  float*    dbc   = (float*)   alloc((size_t)ROWS * DBC_W * 4);
  _Float16* dbch  = (_Float16*)alloc((size_t)ROWS * DBC_W * 2);
  float*    delta = (float*)   alloc((size_t)ROWS * D_INNER * 4);
  _Float16* yzh   = (_Float16*)alloc((size_t)ROWS * D_INNER * 2);
  float*    Aneg  = (float*)   alloc((size_t)D_INNER * D_STATE * 4);

  auto cvt = [&](const float* s, _Float16* d, int n) {
    cvt_f32_f16_kernel<<<(n + 255) / 256, 256, 0, stream>>>(s, d, n);
  };
  cvt(x,     xh,    ROWS * D_MODEL);
  cvt(W_in,  Winh,  2 * D_INNER * D_MODEL);
  cvt(W_x,   Wxh,   DBC_W * D_INNER);
  cvt(W_dt,  Wdth,  D_INNER * DT_RANK);
  cvt(W_out, Wouth, D_MODEL * D_INNER);
  aneg_kernel<<<(D_INNER * D_STATE + 255) / 256, 256, 0, stream>>>(
      A_log, Aneg, D_INNER * D_STATE);

  // 1) in_proj: xz(2048x4096) = x(2048x1024) * W_in^T ; 32x64 tile/wave
  {
    const int M = ROWS, N = 2 * D_INNER, K = D_MODEL;
    const int waves = (M / 32) * (N / 64);               // 4096
    gemm_wmma_kernel<2, 4, 0><<<waves / 8, 256, 0, stream>>>(
        xh, Winh, xzb, nullptr, M, N, K, D_MODEL, D_MODEL, N);
  }
  // 2) depthwise conv + SiLU -> xc (f32 + f16)
  conv_silu_kernel<<<(ROWS * D_INNER + 255) / 256, 256, 0, stream>>>(
      xzb, W_conv, b_conv, xcb, xch);
  // 3) x_proj: dbc(2048x96) = xc * W_x^T ; 32x48 tile/wave
  {
    const int M = ROWS, N = DBC_W, K = D_INNER;
    const int waves = (M / 32) * (N / 48);               // 128
    gemm_wmma_kernel<2, 3, 0><<<waves / 8, 256, 0, stream>>>(
        xch, Wxh, dbc, nullptr, M, N, K, D_INNER, D_INNER, N);
  }
  cvt(dbc, dbch, ROWS * DBC_W);
  // 4) dt_proj + softplus: delta(2048x2048) ; A = dbc[:, :64] (lda=96), K=64
  {
    const int M = ROWS, N = D_INNER, K = DT_RANK;
    const int waves = (M / 32) * (N / 64);               // 2048
    gemm_wmma_kernel<2, 4, 1><<<waves / 8, 256, 0, stream>>>(
        dbch, Wdth, delta, b_dt, M, N, K, DBC_W, DT_RANK, N);
  }
  // 5) SSM streaming update: h_new -> d_out tail, yz (f16)
  ssm_kernel<<<(ROWS * D_INNER + 255) / 256, 256, 0, stream>>>(
      xzb, xcb, dbc, delta, Aneg, D_par, h, h_newp, yzh);
  // 6) out_proj: out(2048x1024) = yz * W_out^T ; 32x64 tile/wave
  {
    const int M = ROWS, N = D_MODEL, K = D_INNER;
    const int waves = (M / 32) * (N / 64);               // 1024
    gemm_wmma_kernel<2, 4, 0><<<waves / 8, 256, 0, stream>>>(
        yzh, Wouth, out_p, nullptr, M, N, K, D_INNER, D_INNER, N);
  }
}
